// MessageFunction_32366873543032
// MI455X (gfx1250) — matrix operand order repaired and checked by reading, not compiled, not run on hardware
//
#include <hip/hip_runtime.h>
#include <hip/hip_bf16.h>
#include <stdint.h>

// Problem sizes (fixed by the reference)
#define B_SZ   32
#define F_K    512          // K per source matrix
#define K_TOT  1024         // fused K = F_EDGE + F_NODE
#define M_SZ   1024
#define N_SZ   4096

// Tiling
#define MT     128
#define NT     128
#define KT     32
#define PH_STEPS (F_K / KT)  // 16 K-steps per source phase

// Padded LDS row lengths (bf16 elements). 40*2 = 80 bytes = 5*16B -> every row 16B aligned,
// and the 20-dword stride spreads accesses across the 64 LDS banks.
#define LDA 40
#define LDB 40

typedef __attribute__((ext_vector_type(16))) __bf16 bf16x16;
typedef __attribute__((ext_vector_type(8)))  __bf16 bf16x8;
typedef __attribute__((ext_vector_type(2)))  __bf16 bf16x2;
typedef __attribute__((ext_vector_type(8)))  float  f32x8;
typedef __attribute__((ext_vector_type(4)))  float  f32x4;

// Pack two f32 -> two bf16 (RNE) in one dword.
// The clang builtin is not declared for gfx1250 on this toolchain (probe: round 2 fell
// back to the scalar path), but the VALU instruction exists on CDNA4+ — emit it directly.
static __device__ __forceinline__ uint32_t pack2_bf16(float lo, float hi) {
#if __has_builtin(__builtin_amdgcn_cvt_pk_bf16_f32)
  union { bf16x2 v; uint32_t u; } c;
  c.v = __builtin_amdgcn_cvt_pk_bf16_f32(lo, hi);
  return c.u;
#else
  uint32_t r;
  asm("v_cvt_pk_bf16_f32 %0, %1, %2" : "=v"(r) : "v"(lo), "v"(hi));
  return r;
#endif
}

__global__ __launch_bounds__(256)
void msg_gemm_bf16(const float* __restrict__ h_w,
                   const float* __restrict__ e_vw,
                   const float* __restrict__ W_e,
                   const float* __restrict__ b_e,
                   const float* __restrict__ W_h,
                   const float* __restrict__ b_h,
                   float* __restrict__ out)
{
  __shared__ __align__(16) uint16_t As[2][MT * LDA];   // bf16 bits, [m][k]
  __shared__ __align__(16) uint16_t Bs[2][NT * LDB];   // bf16 bits, [n][k] (transposed)
  __shared__ float biasS[MT];

  const int tid  = threadIdx.x;
  const int lane = tid & 31;
  const int wid  = tid >> 5;      // 0..7
  const int wm   = wid & 1;       // wave row   (64 M each)
  const int wn   = wid >> 1;      // wave col   (32 N each)

  const int n0 = blockIdx.x * NT;
  const int m0 = blockIdx.y * MT;
  const int b  = blockIdx.z;

  if (tid < MT) biasS[tid] = b_e[m0 + tid] + b_h[m0 + tid];

  // ---- global->reg staging coordinates ----
  // A: each thread loads 16 contiguous floats of one W row (row = tid&127, k half = tid>>7)
  const int a_row = tid & 127;
  const int a_kh  = (tid >> 7) * 16;
  // B: each thread loads a 4(k) x 4(n) micro-tile of X and transposes it in registers
  const int b_nc  = (tid & 31) * 4;   // 0..124
  const int b_kb  = (tid >> 5) * 4;   // 0..28

  const size_t x_batch = (size_t)b * F_K * N_SZ;

  // Phase-invariant per-thread base pointers (phase 0: edge GEMM, phase 1: node GEMM)
  const float* aBase0 = W_e + (size_t)(m0 + a_row) * F_K + a_kh;
  const float* aBase1 = W_h + (size_t)(m0 + a_row) * F_K + a_kh;
  const float* bBase0 = e_vw + x_batch + (size_t)b_kb * N_SZ + (n0 + b_nc);
  const float* bBase1 = h_w  + x_batch + (size_t)b_kb * N_SZ + (n0 + b_nc);

  f32x8 acc[4][2];
#pragma unroll
  for (int i = 0; i < 4; ++i)
#pragma unroll
    for (int j = 0; j < 2; ++j) acc[i][j] = (f32x8)0.0f;

  float a_reg[16];
  float b_reg[4][4];

  auto load_tile = [&](const float* aB, const float* bB, int step) {
    const float* ap = aB + step * KT;
#pragma unroll
    for (int i = 0; i < 4; ++i) {
      f32x4 v = *(const f32x4*)(ap + i * 4);
#pragma unroll
      for (int e = 0; e < 4; ++e) a_reg[i * 4 + e] = v[e];
    }
    const float* bp = bB + (size_t)step * KT * N_SZ;
#pragma unroll
    for (int r = 0; r < 4; ++r) {
      f32x4 v = *(const f32x4*)(bp + (size_t)r * N_SZ);
#pragma unroll
      for (int e = 0; e < 4; ++e) b_reg[r][e] = v[e];
    }
  };

  auto store_tile = [&](int buf) {
    // A: 16 floats -> 8 packed dwords -> two explicit ds_store_b128, row layout [m][k]
    uint32_t pk[8];
#pragma unroll
    for (int i = 0; i < 8; ++i) pk[i] = pack2_bf16(a_reg[2 * i], a_reg[2 * i + 1]);
    uint32_t* adst = (uint32_t*)&As[buf][a_row * LDA + a_kh];
    uint4 q0; q0.x = pk[0]; q0.y = pk[1]; q0.z = pk[2]; q0.w = pk[3];
    uint4 q1; q1.x = pk[4]; q1.y = pk[5]; q1.z = pk[6]; q1.w = pk[7];
    *(uint4*)(adst)     = q0;
    *(uint4*)(adst + 4) = q1;

    // B: register transpose 4x4, store k-contiguous runs into [n][k]
#pragma unroll
    for (int j = 0; j < 4; ++j) {
      uint2 p;
      p.x = pack2_bf16(b_reg[0][j], b_reg[1][j]);
      p.y = pack2_bf16(b_reg[2][j], b_reg[3][j]);
      *(uint2*)&Bs[buf][(b_nc + j) * LDB + b_kb] = p;
    }
  };

  auto compute = [&](int buf) {
    // Fragment layouts per CDNA5 ISA 7.12.2 (16-bit A 16x32, 16-bit B 32x16).
    const int half = (lane >> 4);          // 0: lanes 0-15, 1: lanes 16-31
    const int l16  = lane & 15;
    const int a_c0 = half ? 8 : 0;         // A: first k-chunk
    const int b_k0 = half ? 8 : 0;         // B: k base

    bf16x16 bfrag[2];
#pragma unroll
    for (int j = 0; j < 2; ++j) {
      const int n = wn * 32 + j * 16 + l16;
      const __bf16* p = (const __bf16*)&Bs[buf][n * LDB + b_k0];
      bf16x8 lo = *(const bf16x8*)(p);
      bf16x8 hi = *(const bf16x8*)(p + 16);
      bfrag[j] = __builtin_shufflevector(lo, hi, 0,1,2,3,4,5,6,7,8,9,10,11,12,13,14,15);
    }

#pragma unroll
    for (int i = 0; i < 4; ++i) {
      const int m = wm * 64 + i * 16 + l16;
      const __bf16* p = (const __bf16*)&As[buf][m * LDA];
      bf16x8 lo = *(const bf16x8*)(p + a_c0);
      bf16x8 hi = *(const bf16x8*)(p + a_c0 + 16);
      bf16x16 afrag = __builtin_shufflevector(lo, hi, 0,1,2,3,4,5,6,7,8,9,10,11,12,13,14,15);
#pragma unroll
      for (int j = 0; j < 2; ++j) {
        acc[i][j] = __builtin_amdgcn_wmma_f32_16x16x32_bf16(
            false, afrag, false, bfrag[j], (short)0, acc[i][j], false, false);
      }
    }
  };

  // ---- software pipeline: double-buffered LDS, two source phases ----
  load_tile(aBase0, bBase0, 0);
  store_tile(0);
  __syncthreads();

  // Phase 0 (W_e x e_vw): compute steps 0..14, prefetching step+1 from phase-0 sources
  for (int s = 0; s < PH_STEPS - 1; ++s) {
    load_tile(aBase0, bBase0, s + 1);
    compute(s & 1);
    store_tile((s + 1) & 1);
    __syncthreads();
  }
  // Boundary step (global kt = 15): prefetch first tile of phase 1
  load_tile(aBase1, bBase1, 0);
  compute((PH_STEPS - 1) & 1);
  store_tile(PH_STEPS & 1);
  __syncthreads();

  // Phase 1 (W_h x h_w): global kt = 16 + s, same parity as s
  for (int s = 0; s < PH_STEPS - 1; ++s) {
    load_tile(aBase1, bBase1, s + 1);
    compute(s & 1);
    store_tile((s + 1) & 1);
    __syncthreads();
  }
  compute((PH_STEPS - 1) & 1);   // last tile, no prefetch

  // ---- epilogue: bias + store (C/D layout: elem r -> M=r (lanes<16) / M=8+r) ----
  const int l16  = lane & 15;
  const int moff = (lane >> 4) * 8;
#pragma unroll
  for (int i = 0; i < 4; ++i) {
    const int mloc = wm * 64 + i * 16 + moff;
#pragma unroll
    for (int r = 0; r < 8; ++r) {
      const float bias = biasS[mloc + r];
      const size_t mrow = (size_t)(m0 + mloc + r) * N_SZ;
#pragma unroll
      for (int j = 0; j < 2; ++j) {
        const int n = n0 + wn * 32 + j * 16 + l16;
        out[(size_t)b * M_SZ * N_SZ + mrow + n] = acc[i][j][r] + bias;
      }
    }
  }
}

extern "C" void kernel_launch(void* const* d_in, const int* in_sizes, int n_in,
                              void* d_out, int out_size, void* d_ws, size_t ws_size,
                              hipStream_t stream) {
  (void)in_sizes; (void)n_in; (void)d_ws; (void)ws_size; (void)out_size;
  // inputs: 0:h_v (unused), 1:h_w, 2:e_vw, 3:W_e, 4:b_e, 5:W_h, 6:b_h
  const float* h_w  = (const float*)d_in[1];
  const float* e_vw = (const float*)d_in[2];
  const float* W_e  = (const float*)d_in[3];
  const float* b_e  = (const float*)d_in[4];
  const float* W_h  = (const float*)d_in[5];
  const float* b_h  = (const float*)d_in[6];
  float* out = (float*)d_out;

  dim3 grid(N_SZ / NT, M_SZ / MT, B_SZ);  // (32, 8, 32)
  msg_gemm_bf16<<<grid, 256, 0, stream>>>(h_w, e_vw, W_e, b_e, W_h, b_h, out);
}
